// VoxelWideResBackBone8x_59030030516569
// MI455X (gfx1250) — matrix operand order repaired and checked by reading, not compiled, not run on hardware
//
#include <hip/hip_runtime.h>
#include <hip/hip_bf16.h>

typedef __attribute__((ext_vector_type(16))) __bf16 v16bf;
typedef __attribute__((ext_vector_type(8)))  float  v8f;

// ---------------------------------------------------------------------------
// zero-fill
// ---------------------------------------------------------------------------
__global__ void zero_f32(float* __restrict__ p, long n) {
  long i = blockIdx.x * (long)blockDim.x + threadIdx.x;
  if (i < n) p[i] = 0.0f;
}

// ---------------------------------------------------------------------------
// Pack f32 weights [taps][Cin][Cout] into WMMA-B bf16 fragments:
//   frag index f = (k*ntiles + nt)*kch + kc ; element  [lane][j]
//   value = w[k][kc*32 + (lane>=16?16:0) + j][nt*16 + (lane&15)]  (0 if K>=Cin)
// ---------------------------------------------------------------------------
__global__ void pack_w(const float* __restrict__ w, __bf16* __restrict__ out,
                       int taps, int cin, int cout, long total) {
  long e = blockIdx.x * (long)blockDim.x + threadIdx.x;
  if (e >= total) return;
  const int kch    = (cin + 31) / 32;
  const int ntiles = cout / 16;
  long idx = e;
  const int j    = (int)(idx & 15);  idx >>= 4;
  const int lane = (int)(idx & 31);  idx >>= 5;
  const int kc   = (int)(idx % kch); idx /= kch;
  const int nt   = (int)(idx % ntiles); idx /= ntiles;
  const int k    = (int)idx;
  const int kdim = kc * 32 + ((lane >= 16) ? 16 : 0) + j;
  const int col  = nt * 16 + (lane & 15);
  float v = (kdim < cin) ? w[((long)k * cin + kdim) * cout + col] : 0.0f;
  out[e] = (__bf16)v;
}

// ---------------------------------------------------------------------------
// Gather -> WMMA(bf16) -> masked atomic scatter for one kernel tap.
// Block = 1 wave. Tile: 16 pairs x (16*NT) channels.
// grid = (ceil(P/16), COUT/(16*NT), taps)
//
// A (16x32 bf16): lane l -> row (l&15); elem j: K = koff + j,
//   koff = (l>=16)?16:0  -> two 8-float runs, loaded as 4x float4.
// B: pre-packed fragments, one 32B load per lane.
// D (16x16 f32): lane l -> col (l&15); vgpr v -> row v + (l>=16?8:0).
// ---------------------------------------------------------------------------
template <int CIN, int COUT, int NT>
__global__ void tap_gemm(const float* __restrict__ x,
                         const __bf16* __restrict__ wp,
                         const int*   __restrict__ rb_in,
                         const int*   __restrict__ rb_out,
                         const float* __restrict__ rb_mask,
                         float* __restrict__ acc, int P) {
  constexpr int KCH    = (CIN + 31) / 32;
  constexpr int NTILES = COUT / 16;
  const int lane = threadIdx.x & 31;
  const int k    = blockIdx.z;
  const int m0   = blockIdx.x * 16;
  const int nt0  = blockIdx.y * NT;
  const int l15  = lane & 15;
  const bool hi  = lane >= 16;

  // lanes 0..15 hold metadata for pair rows m0+lane
  int rowIdx = 0, outIdx = 0;
  float mk = 0.0f;
  if (lane < 16) {
    int m = m0 + lane;
    if (m < P) {
      long q = (long)k * P + m;
      rowIdx = rb_in[q];
      outIdx = rb_out[q];
      mk     = rb_mask[q];
    }
  }
  const int myRow = __shfl(rowIdx, l15, 32);
  const float* __restrict__ xrow = x + (long)myRow * CIN;
  const v16bf* __restrict__ bp = (const v16bf*)wp;

  v8f d[NT];
#pragma unroll
  for (int t = 0; t < NT; ++t) d[t] = (v8f)0.0f;

  const int koff = hi ? 16 : 0;

#pragma unroll
  for (int kc = 0; kc < KCH; ++kc) {
    // ---- A fragment ----
    v16bf a;
    if constexpr (CIN == 4) {
#pragma unroll
      for (int j = 0; j < 16; ++j) a[j] = (__bf16)0.0f;
      if (!hi) {
        float4 r0 = *(const float4*)xrow;
        a[0] = (__bf16)r0.x; a[1] = (__bf16)r0.y;
        a[2] = (__bf16)r0.z; a[3] = (__bf16)r0.w;
      }
    } else {
      const float4* xv1 = (const float4*)(xrow + kc * 32 + koff);
      const float4* xv2 = (const float4*)(xrow + kc * 32 + 8 + koff);
      float4 r0 = xv1[0], r1 = xv1[1];
      float4 r2 = xv2[0], r3 = xv2[1];
      a[0]  = (__bf16)r0.x; a[1]  = (__bf16)r0.y;
      a[2]  = (__bf16)r0.z; a[3]  = (__bf16)r0.w;
      a[4]  = (__bf16)r1.x; a[5]  = (__bf16)r1.y;
      a[6]  = (__bf16)r1.z; a[7]  = (__bf16)r1.w;
      a[8]  = (__bf16)r2.x; a[9]  = (__bf16)r2.y;
      a[10] = (__bf16)r2.z; a[11] = (__bf16)r2.w;
      a[12] = (__bf16)r3.x; a[13] = (__bf16)r3.y;
      a[14] = (__bf16)r3.z; a[15] = (__bf16)r3.w;
    }
    // ---- NT WMMAs against pre-packed B fragments ----
#pragma unroll
    for (int t = 0; t < NT; ++t) {
      const long f = ((long)(k * NTILES + (nt0 + t)) * KCH + kc);
      v16bf b = bp[f * 32 + lane];
      d[t] = __builtin_amdgcn_wmma_f32_16x16x32_bf16(
          false, a, false, b, (short)0, d[t], false, false);
    }
  }

  // ---- masked atomic scatter ----
#pragma unroll
  for (int v = 0; v < 8; ++v) {
    const int r = v + (hi ? 8 : 0);
    const int m = m0 + r;
    const int   om = __shfl(outIdx, r, 32);
    const float mv = __shfl(mk, r, 32);
    if (m < P && mv != 0.0f) {
#pragma unroll
      for (int t = 0; t < NT; ++t) {
        atomicAdd(&acc[(long)om * COUT + (nt0 + t) * 16 + l15], d[t][v] * mv);
      }
    }
  }
}

// ---------------------------------------------------------------------------
// epilogue: y = relu( (acc + bias - mean)*rsqrt(var+eps)*gamma + beta + resid )
// ---------------------------------------------------------------------------
__global__ void epilogue(const float* __restrict__ acc,
                         const float* __restrict__ bias,    // nullable
                         const float* __restrict__ beta,
                         const float* __restrict__ gamma,
                         const float* __restrict__ mean,
                         const float* __restrict__ var,
                         const float* __restrict__ resid,   // nullable
                         float* __restrict__ dst,
                         long n, int C) {
  long i = blockIdx.x * (long)blockDim.x + threadIdx.x;
  if (i >= n) return;
  int c = (int)(i % C);
  float v = acc[i];
  if (bias) v += bias[c];
  v = (v - mean[c]) * rsqrtf(var[c] + 1e-3f) * gamma[c] + beta[c];
  if (resid) v += resid[i];
  dst[i] = fmaxf(v, 0.0f);
}

// ---------------------------------------------------------------------------
// host-side helpers
// ---------------------------------------------------------------------------
static inline long pack_elems(long taps, long cin, long cout) {
  long kch = (cin + 31) / 32;
  return taps * (cout / 16) * kch * 512;  // 32 lanes * 16 elems per fragment
}

static void run_conv(hipStream_t s, const float* x, const float* w,
                     const int* ri, const int* ro, const float* rm,
                     int P, int taps, int Cin, int Cout, int nOut,
                     float* acc, __bf16* packbuf) {
  // pack weights -> bf16 fragments
  long pe = pack_elems(taps, Cin, Cout);
  pack_w<<<dim3((unsigned)((pe + 255) / 256)), 256, 0, s>>>(
      w, packbuf, taps, Cin, Cout, pe);
  // zero accumulator
  long total = (long)nOut * Cout;
  zero_f32<<<dim3((unsigned)((total + 255) / 256)), 256, 0, s>>>(acc, total);
  // gemm
  dim3 grid((P + 15) / 16, (Cout >= 64) ? (Cout / 64) : 1, taps);
#define GEMM(CI, CO, NT) \
  tap_gemm<CI, CO, NT><<<grid, 32, 0, s>>>(x, packbuf, ri, ro, rm, acc, P)
  if      (Cin == 4   && Cout == 32)  GEMM(4, 32, 2);
  else if (Cin == 32  && Cout == 32)  GEMM(32, 32, 2);
  else if (Cin == 32  && Cout == 64)  GEMM(32, 64, 4);
  else if (Cin == 64  && Cout == 64)  GEMM(64, 64, 4);
  else if (Cin == 64  && Cout == 128) GEMM(64, 128, 4);
  else if (Cin == 128 && Cout == 128) GEMM(128, 128, 4);
  else if (Cin == 128 && Cout == 256) GEMM(128, 256, 4);
  else if (Cin == 256 && Cout == 256) GEMM(256, 256, 4);
#undef GEMM
}

static inline void run_epi(hipStream_t s, const float* acc, const float* bias,
                           const float* beta, const float* gamma,
                           const float* mean, const float* var,
                           const float* resid, float* dst, int nOut, int C) {
  long n = (long)nOut * C;
  epilogue<<<dim3((unsigned)((n + 255) / 256)), 256, 0, s>>>(
      acc, bias, beta, gamma, mean, var, resid, dst, n, C);
}

// Flattened input indices (jax tree order: dict keys sorted, lists in order)
// params: down2(0-4) down3(5-9) down4(10-14) inp(15-19) out(20-24)
//         res1[0](25-36) res1[1](37-48) res2(49-72) res3(73-96) res4(97-120)
//   per down/inp/out: bn.beta, bn.gamma, bn.mean, bn.var, w
//   per block: b1,b2, bn1.{beta,gamma,mean,var}, bn2.{beta,gamma,mean,var}, w1,w2
// rulebooks: sp2(121-124) sp3(125-128) sp4(129-132) spout(133-136)
//            sub1(137-140) sub2(141-144) sub3(145-148) sub4(149-152)
//   per rb: inp, mask, n_out, out
// voxel_coords: 153   voxel_features: 154

extern "C" void kernel_launch(void* const* d_in, const int* in_sizes, int n_in,
                              void* d_out, int out_size, void* d_ws,
                              size_t ws_size, hipStream_t stream) {
  auto F = [&](int i) { return (const float*)d_in[i]; };
  auto I = [&](int i) { return (const int*)d_in[i]; };

  const int n1 = in_sizes[137] / 27;  // sub1 pairs-per-tap == voxel count L1
  const int n2 = in_sizes[141] / 27;
  const int n3 = in_sizes[145] / 27;
  const int n4 = in_sizes[149] / 27;
  const int nout = out_size / 256;
  const int Psp2 = in_sizes[121] / 27;
  const int Psp3 = in_sizes[125] / 27;
  const int Psp4 = in_sizes[129] / 27;
  const int Pspo = in_sizes[133] / 3;

  // carve workspace
  float* ws = (float*)d_ws;
  float* x1a = ws;            ws += (long)n1 * 32;
  float* x1b = ws;            ws += (long)n1 * 32;
  float* x2a = ws;            ws += (long)n2 * 64;
  float* x2b = ws;            ws += (long)n2 * 64;
  float* x3a = ws;            ws += (long)n3 * 128;
  float* x3b = ws;            ws += (long)n3 * 128;
  float* x4a = ws;            ws += (long)n4 * 256;
  float* x4b = ws;            ws += (long)n4 * 256;
  long accMax = (long)n1 * 32;
  if ((long)n2 * 64  > accMax) accMax = (long)n2 * 64;
  if ((long)n3 * 128 > accMax) accMax = (long)n3 * 128;
  if ((long)n4 * 256 > accMax) accMax = (long)n4 * 256;
  if ((long)nout * 256 > accMax) accMax = (long)nout * 256;
  float* acc = ws;            ws += accMax;
  __bf16* packbuf = (__bf16*)ws;  // max pack = res4: 27*16*8*512 bf16 (~3.5MB)

  const float* vf = F(154);

  // ---- input layer: SubMConv 4->32 + BN + ReLU ----
  run_conv(stream, vf, F(19), I(137), I(140), F(138), n1, 27, 4, 32, n1, acc, packbuf);
  run_epi(stream, acc, nullptr, F(15), F(16), F(17), F(18), nullptr, x1a, n1, 32);

  struct Lvl {
    int base0, base1;
    const int *ri, *ro; const float* rm;
    int P, C, n;
    float *xa, *xb;
  };
  const Lvl lv1 = {25, 37, I(137), I(140), F(138), n1, 32, n1, x1a, x1b};
  const Lvl lv2 = {49, 61, I(141), I(144), F(142), n2, 64, n2, x2a, x2b};
  const Lvl lv3 = {73, 85, I(145), I(148), F(146), n3, 128, n3, x3a, x3b};
  const Lvl lv4 = {97, 109, I(149), I(152), F(150), n4, 256, n4, x4a, x4b};

  auto res_blocks = [&](const Lvl& L) {
    const int bases[2] = {L.base0, L.base1};
    for (int bi = 0; bi < 2; ++bi) {
      int B = bases[bi];
      run_conv(stream, L.xa, F(B + 10), L.ri, L.ro, L.rm, L.P, 27, L.C, L.C,
               L.n, acc, packbuf);
      run_epi(stream, acc, F(B + 0), F(B + 2), F(B + 3), F(B + 4), F(B + 5),
              nullptr, L.xb, L.n, L.C);
      run_conv(stream, L.xb, F(B + 11), L.ri, L.ro, L.rm, L.P, 27, L.C, L.C,
               L.n, acc, packbuf);
      run_epi(stream, acc, F(B + 1), F(B + 6), F(B + 7), F(B + 8), F(B + 9),
              L.xa, L.xa, L.n, L.C);
    }
  };

  res_blocks(lv1);

  run_conv(stream, x1a, F(4), I(121), I(124), F(122), Psp2, 27, 32, 64, n2, acc, packbuf);
  run_epi(stream, acc, nullptr, F(0), F(1), F(2), F(3), nullptr, x2a, n2, 64);
  res_blocks(lv2);

  run_conv(stream, x2a, F(9), I(125), I(128), F(126), Psp3, 27, 64, 128, n3, acc, packbuf);
  run_epi(stream, acc, nullptr, F(5), F(6), F(7), F(8), nullptr, x3a, n3, 128);
  res_blocks(lv3);

  run_conv(stream, x3a, F(14), I(129), I(132), F(130), Psp4, 27, 128, 256, n4, acc, packbuf);
  run_epi(stream, acc, nullptr, F(10), F(11), F(12), F(13), nullptr, x4a, n4, 256);
  res_blocks(lv4);

  run_conv(stream, x4a, F(24), I(133), I(136), F(134), Pspo, 3, 256, 256, nout, acc, packbuf);
  run_epi(stream, acc, nullptr, F(20), F(21), F(22), F(23), nullptr,
          (float*)d_out, nout, 256);
}
